// MultiHeadAttention_59261958750394
// MI455X (gfx1250) — matrix operand order repaired and checked by reading, not compiled, not run on hardware
//
#include <hip/hip_runtime.h>

// ---------------- types ----------------
typedef __attribute__((ext_vector_type(16))) __bf16 v16bf;
typedef __attribute__((ext_vector_type(8)))  float  v8f;

static constexpr int D_MODEL  = 1024;
static constexpr int NUM_HEADS = 16;
static constexpr int HEAD_DIM = 64;
static constexpr int BATCH    = 2;
static constexpr int SEQ      = 2048;

__device__ __forceinline__ float bf2f(__bf16 b) {
    unsigned short s = __builtin_bit_cast(unsigned short, b);
    unsigned u = ((unsigned)s) << 16;
    return __builtin_bit_cast(float, u);
}
__device__ __forceinline__ __bf16 f2bf(float f) {
    unsigned u = __builtin_bit_cast(unsigned, f);
    u += 0x7fffu + ((u >> 16) & 1u);   // round-to-nearest-even
    return __builtin_bit_cast(__bf16, (unsigned short)(u >> 16));
}
__device__ __forceinline__ v8f wmma_bf16(v16bf a, v16bf b, v8f c) {
    return __builtin_amdgcn_wmma_f32_16x16x32_bf16(
        false, a, false, b, (short)0, c, false, false);
}

// ---------------- fp32 -> bf16 convert ----------------
__global__ void cvt_f32_bf16(const float* __restrict__ src,
                             __bf16* __restrict__ dst, size_t n) {
    size_t i = (size_t)blockIdx.x * blockDim.x + threadIdx.x;
    if (i < n) dst[i] = f2bf(src[i]);
}

// ---------------- tiled bf16 WMMA GEMM:  C = A(MxK) * Bw(NxK)^T ----------
// Double-buffered pipeline: CDNA5 async global->LDS DMA (ASYNCcnt) stages
// tile i+1 while 16 WMMAs consume tile i. Async ops bypass VGPRs entirely.
// MODE 0: C = float, row-major [M][N]
// MODE 1: C = bf16 head-major [B, H, T, 64]  (N = H*64, M = B*T)
template <int MODE>
__global__ __launch_bounds__(256) void gemm_bf16_nt(
    const __bf16* __restrict__ A, const __bf16* __restrict__ Bw,
    void* __restrict__ Cout, int M, int N, int K, int Tdim, int Hh)
{
    constexpr int BK   = 64;                 // k-span per pipeline stage
    constexpr int LDSS = 72;                 // 64 + 8 pad (bf16 elems)
    __shared__ __bf16 As[2][128 * LDSS];     // 2 x 18KB
    __shared__ __bf16 Bs[2][128 * LDSS];     // 2 x 18KB

    const int tid  = threadIdx.x;
    const int lane = tid & 31;
    const int wave = tid >> 5;
    const int m0 = blockIdx.y * 128;
    const int n0 = blockIdx.x * 128;
    const int wm = (wave >> 1) * 32;         // wave row base (4 wave-rows)
    const int wn = (wave & 1) * 64;          // wave col base (2 wave-cols)
    const int lr   = lane & 15;
    const int half = lane >> 4;              // 0: lanes 0-15, 1: lanes 16-31

    // per-thread staging: 128x64 tile, 32 bf16 = 64B per thread per matrix
    const int r = tid >> 1;
    const int c = (tid & 1) * 32;
    const __bf16* gaBase = A  + (size_t)(m0 + r) * K + c;
    const __bf16* gbBase = Bw + (size_t)(n0 + r) * K + c;
    const unsigned laB[2] = { (unsigned)(uintptr_t)(&As[0][r * LDSS + c]),
                              (unsigned)(uintptr_t)(&As[1][r * LDSS + c]) };
    const unsigned lbB[2] = { (unsigned)(uintptr_t)(&Bs[0][r * LDSS + c]),
                              (unsigned)(uintptr_t)(&Bs[1][r * LDSS + c]) };

    // async DMA of one 128x64 A tile + one 128x64 B tile into buffer `buf`
    auto dma = [&](int k0, int buf) {
        const __bf16* ga = gaBase + k0;
        const __bf16* gb = gbBase + k0;
        asm volatile(
            "global_load_async_to_lds_b128 %0, %2, off\n\t"
            "global_load_async_to_lds_b128 %0, %2, off offset:16\n\t"
            "global_load_async_to_lds_b128 %0, %2, off offset:32\n\t"
            "global_load_async_to_lds_b128 %0, %2, off offset:48\n\t"
            "global_load_async_to_lds_b128 %1, %3, off\n\t"
            "global_load_async_to_lds_b128 %1, %3, off offset:16\n\t"
            "global_load_async_to_lds_b128 %1, %3, off offset:32\n\t"
            "global_load_async_to_lds_b128 %1, %3, off offset:48"
            :
            : "v"(laB[buf]), "v"(lbB[buf]), "v"(ga), "v"(gb)
            : "memory");
    };

    v8f acc[2][4] = {};

    // prologue: stage 0
    dma(0, 0);
    asm volatile("s_wait_asynccnt 0x0" ::: "memory");
    __syncthreads();

    for (int k0 = 0; k0 < K; k0 += BK) {
        const int buf = (k0 / BK) & 1;
        if (k0 + BK < K) {
            dma(k0 + BK, buf ^ 1);           // overlap DMA with WMMA below
            if (k0 + 2 * BK < K) {
                __builtin_prefetch(gaBase + k0 + 2 * BK, 0, 0);
                __builtin_prefetch(gbBase + k0 + 2 * BK, 0, 0);
            }
        }

#pragma unroll
        for (int s = 0; s < BK; s += 32) {
            // A fragments: lane<16 -> k {s..s+7, s+16..s+23}; lanes 16-31 complement
            v16bf af[2];
#pragma unroll
            for (int i = 0; i < 2; ++i) {
                const __bf16* p = &As[buf][(wm + i * 16 + lr) * LDSS + s + half * 8];
                ((uint4*)&af[i])[0] = *(const uint4*)p;
                ((uint4*)&af[i])[1] = *(const uint4*)(p + 16);
            }
            // B fragments: lane n holds column n; lane<16 -> k {s..s+7, s+16..s+23}; lanes 16-31 complement
            v16bf bfr[4];
#pragma unroll
            for (int j = 0; j < 4; ++j) {
                const __bf16* p = &Bs[buf][(wn + j * 16 + lr) * LDSS + s + half * 8];
                ((uint4*)&bfr[j])[0] = *(const uint4*)p;
                ((uint4*)&bfr[j])[1] = *(const uint4*)(p + 16);
            }
#pragma unroll
            for (int i = 0; i < 2; ++i)
#pragma unroll
                for (int j = 0; j < 4; ++j)
                    acc[i][j] = wmma_bf16(af[i], bfr[j], acc[i][j]);
        }

        asm volatile("s_wait_asynccnt 0x0" ::: "memory");
        __syncthreads();
    }

    // store: C/D layout -> VGPR v holds row (half? v+8 : v), col = lr
    for (int i = 0; i < 2; ++i)
        for (int j = 0; j < 4; ++j)
            for (int v = 0; v < 8; ++v) {
                const int m = m0 + wm + i * 16 + half * 8 + v;
                const int n = n0 + wn + j * 16 + lr;
                const float val = acc[i][j][v];
                if (MODE == 0) {
                    ((float*)Cout)[(size_t)m * N + n] = val;
                } else {
                    const int b = m / Tdim, t = m % Tdim;
                    const int h = n >> 6,  d = n & 63;
                    ((__bf16*)Cout)[(((size_t)(b * Hh + h)) * Tdim + t) * 64 + d] =
                        f2bf(val);
                }
            }
}

// ---------------- RoPE (in-place, bf16, layout [B*H, T, 64]) --------------
__global__ void rope_kernel(__bf16* __restrict__ X, int T, size_t npairs) {
    size_t i = (size_t)blockIdx.x * blockDim.x + threadIdx.x;
    if (i >= npairs) return;
    const int d2 = (int)(i & 31);            // pair index 0..31
    const size_t rt = i >> 5;                // bh*T + t
    const int t = (int)(rt % (size_t)T);
    const size_t base = rt * 64 + 2 * (size_t)d2;
    // inv_freq = 10000^(-2*d2/64) = exp2(-(2*d2/64) * log2(10000))
    const float inv_freq = exp2f(-((float)(2 * d2) / 64.0f) * 13.2877123795f);
    const float ang = (float)t * inv_freq;
    float sn, cs;
    __sincosf(ang, &sn, &cs);
    const float e = bf2f(X[base]);
    const float o = bf2f(X[base + 1]);
    X[base]     = f2bf(e * cs - o * sn);
    X[base + 1] = f2bf(e * sn + o * cs);
}

// ---------------- flash attention (causal) --------------------------------
// grid: (T/64 q-chunks, B*H).  block: 128 (4 waves). wave owns 16 q-rows.
__global__ __launch_bounds__(128) void flash_attn_kernel(
    const __bf16* __restrict__ Q, const __bf16* __restrict__ Kt,
    const __bf16* __restrict__ V, __bf16* __restrict__ O, int T, int H)
{
    constexpr int LDSS = 40;
    __shared__ __bf16 Vt[64 * LDSS];         // transposed V tile: [d][k]
    __shared__ __bf16 Pt[4][16 * LDSS];      // per-wave P staging (A layout)

    const int tid  = threadIdx.x;
    const int lane = tid & 31;
    const int wave = tid >> 5;
    const int bh = blockIdx.y;
    const int qc = blockIdx.x;
    const size_t hbase = (size_t)bh * T * 64;
    const int qbase = qc * 64 + wave * 16;
    const int lr   = lane & 15;
    const int half = lane >> 4;

    // Q A-fragments for dh steps {0..31, 32..63}
    v16bf qa[2];
    {
        const __bf16* qrow = Q + hbase + (size_t)(qbase + lr) * 64 + half * 8;
        for (int s = 0; s < 2; ++s) {
            ((uint4*)&qa[s])[0] = *(const uint4*)(qrow + s * 32);
            ((uint4*)&qa[s])[1] = *(const uint4*)(qrow + s * 32 + 16);
        }
    }

    float mrow[8], lsum[8];
    for (int v = 0; v < 8; ++v) { mrow[v] = -1e30f; lsum[v] = 0.0f; }
    v8f oacc[4] = {};

    const int kmax = qc * 64 + 64;           // causal upper bound for this block
    for (int kt = 0; kt < kmax; kt += 32) {
        // cooperative transposed V staging: V[kt..kt+31][0..63] -> Vt[d][k]
        {
            const int r = tid >> 2;
            const int seg = (tid & 3) * 16;
            const __bf16* vrow = V + hbase + (size_t)(kt + r) * 64 + seg;
            __bf16 tmp[16];
            *(uint4*)&tmp[0] = *(const uint4*)vrow;
            *(uint4*)&tmp[8] = *(const uint4*)(vrow + 8);
            for (int cidx = 0; cidx < 16; ++cidx)
                Vt[(seg + cidx) * LDSS + r] = tmp[cidx];
        }
        __syncthreads();

        // S = Q K^T for two 16-wide kv subtiles, scaled + causal-masked
        v8f ssub[2];
        for (int sub = 0; sub < 2; ++sub) {
            v16bf kb[2];
            const __bf16* krow =
                Kt + hbase + (size_t)(kt + sub * 16 + lr) * 64 + half * 8;
            for (int s = 0; s < 2; ++s) {
                ((uint4*)&kb[s])[0] = *(const uint4*)(krow + s * 32);
                ((uint4*)&kb[s])[1] = *(const uint4*)(krow + s * 32 + 16);
            }
            v8f s8 = {};
            s8 = wmma_bf16(qa[0], kb[0], s8);
            s8 = wmma_bf16(qa[1], kb[1], s8);
            for (int v = 0; v < 8; ++v) {
                const int mg = qbase + half * 8 + v;
                const int ng = kt + sub * 16 + lr;
                const float sv = s8[v] * 0.125f;          // 1/sqrt(64)
                s8[v] = (ng <= mg) ? sv : -1e30f;
            }
            ssub[sub] = s8;
        }

        // online softmax (row lives in 16 lanes of one half)
        float alpha[8];
        for (int v = 0; v < 8; ++v) {
            float rmax = fmaxf(ssub[0][v], ssub[1][v]);
            for (int off = 8; off; off >>= 1)
                rmax = fmaxf(rmax, __shfl_xor(rmax, off, 32));
            const float mn = fmaxf(mrow[v], rmax);
            const float a  = __expf(mrow[v] - mn);
            const float p0 = __expf(ssub[0][v] - mn);
            const float p1 = __expf(ssub[1][v] - mn);
            float rs = p0 + p1;
            for (int off = 8; off; off >>= 1)
                rs += __shfl_xor(rs, off, 32);
            lsum[v] = lsum[v] * a + rs;
            mrow[v] = mn;
            alpha[v] = a;
            ssub[0][v] = p0; ssub[1][v] = p1;
        }
        for (int j = 0; j < 4; ++j)
            for (int v = 0; v < 8; ++v) oacc[j][v] *= alpha[v];

        // bounce P (C layout) through LDS into A layout
        __bf16* pw = Pt[wave];
        for (int v = 0; v < 8; ++v) {
            const int row = half * 8 + v;
            pw[row * LDSS + lr]      = f2bf(ssub[0][v]);
            pw[row * LDSS + 16 + lr] = f2bf(ssub[1][v]);
        }
        asm volatile("s_wait_dscnt 0x0" ::: "memory");   // wave-private RAW

        v16bf pa;
        {
            const __bf16* p = pw + lr * LDSS + half * 8;
            ((uint4*)&pa)[0] = *(const uint4*)p;
            ((uint4*)&pa)[1] = *(const uint4*)(p + 16);
        }
        // O += P * V  (V B-fragments from transposed LDS tile)
        for (int j = 0; j < 4; ++j) {
            v16bf vb;
            const __bf16* p = &Vt[(j * 16 + lr) * LDSS + half * 8];
            ((uint4*)&vb)[0] = *(const uint4*)p;
            ((uint4*)&vb)[1] = *(const uint4*)(p + 16);
            oacc[j] = wmma_bf16(pa, vb, oacc[j]);
        }
        __syncthreads();
    }

    // normalize and store merged-head layout [B, T, H*64] (bf16)
    const int b = bh / H, h = bh % H;
    for (int j = 0; j < 4; ++j)
        for (int v = 0; v < 8; ++v) {
            const int t = qbase + half * 8 + v;
            const int col = h * 64 + j * 16 + lr;
            const float val = oacc[j][v] / lsum[v];
            O[((size_t)(b * T + t)) * (size_t)(H * 64) + col] = f2bf(val);
        }
}

// ---------------- launch ----------------
extern "C" void kernel_launch(void* const* d_in, const int* in_sizes, int n_in,
                              void* d_out, int out_size, void* d_ws, size_t ws_size,
                              hipStream_t stream) {
    const float* x  = (const float*)d_in[0];
    // d_in[1] = mask (bool) -- causality is computed analytically, not read
    const float* Wq = (const float*)d_in[2];
    const float* Wk = (const float*)d_in[3];
    const float* Wv = (const float*)d_in[4];
    const float* Wo = (const float*)d_in[5];
    float* out = (float*)d_out;

    const size_t MB = 1024 * 1024;
    char* w = (char*)d_ws;
    __bf16* xb  = (__bf16*)(w + 0);        // 8 MiB  : x bf16 [B*T, D]
    __bf16* Wqb = (__bf16*)(w + 8  * MB);  // 2 MiB
    __bf16* Wkb = (__bf16*)(w + 10 * MB);  // 2 MiB
    __bf16* Wvb = (__bf16*)(w + 12 * MB);  // 2 MiB
    __bf16* Wob = (__bf16*)(w + 14 * MB);  // 2 MiB
    __bf16* Qb  = (__bf16*)(w + 16 * MB);  // 8 MiB  : [B,H,T,64]
    __bf16* Kb  = (__bf16*)(w + 24 * MB);  // 8 MiB
    __bf16* Vb  = (__bf16*)(w + 32 * MB);  // 8 MiB
    __bf16* Ob  = (__bf16*)(w + 40 * MB);  // 8 MiB  : [B,T,D]

    const size_t nX = (size_t)BATCH * SEQ * D_MODEL;     // 4,194,304
    const size_t nW = (size_t)D_MODEL * D_MODEL;         // 1,048,576
    const int M = BATCH * SEQ;                           // 4096
    const int N = D_MODEL, K = D_MODEL;

    // 1) convert to bf16
    cvt_f32_bf16<<<(int)((nX + 255) / 256), 256, 0, stream>>>(x, xb, nX);
    cvt_f32_bf16<<<(int)((nW + 255) / 256), 256, 0, stream>>>(Wq, Wqb, nW);
    cvt_f32_bf16<<<(int)((nW + 255) / 256), 256, 0, stream>>>(Wk, Wkb, nW);
    cvt_f32_bf16<<<(int)((nW + 255) / 256), 256, 0, stream>>>(Wv, Wvb, nW);
    cvt_f32_bf16<<<(int)((nW + 255) / 256), 256, 0, stream>>>(Wo, Wob, nW);

    // 2) Q/K/V projections (bf16 WMMA, head-major bf16 output)
    dim3 ggrid(N / 128, M / 128);
    gemm_bf16_nt<1><<<ggrid, 256, 0, stream>>>(xb, Wqb, Qb, M, N, K, SEQ, NUM_HEADS);
    gemm_bf16_nt<1><<<ggrid, 256, 0, stream>>>(xb, Wkb, Kb, M, N, K, SEQ, NUM_HEADS);
    gemm_bf16_nt<1><<<ggrid, 256, 0, stream>>>(xb, Wvb, Vb, M, N, K, SEQ, NUM_HEADS);

    // 3) RoPE on Q and K
    const size_t npairs = (size_t)BATCH * NUM_HEADS * SEQ * (HEAD_DIM / 2);
    rope_kernel<<<(int)((npairs + 255) / 256), 256, 0, stream>>>(Qb, SEQ, npairs);
    rope_kernel<<<(int)((npairs + 255) / 256), 256, 0, stream>>>(Kb, SEQ, npairs);

    // 4) causal flash attention
    dim3 agrid(SEQ / 64, BATCH * NUM_HEADS);
    flash_attn_kernel<<<agrid, 128, 0, stream>>>(Qb, Kb, Vb, Ob, SEQ, NUM_HEADS);

    // 5) output projection -> fp32 d_out
    gemm_bf16_nt<0><<<ggrid, 256, 0, stream>>>(Ob, Wob, out, M, N, K, SEQ, NUM_HEADS);

    (void)in_sizes; (void)n_in; (void)out_size; (void)ws_size;
}
